// SVGAutoencoder_5935644803199
// MI455X (gfx1250) — matrix operand order repaired and checked by reading, not compiled, not run on hardware
//
#include <hip/hip_runtime.h>
#include <hip/hip_bf16.h>

// ---------------------------------------------------------------------------
// Static problem dims (from reference setup_inputs)
// ---------------------------------------------------------------------------
#define BB   8
#define NF   4096
#define DD   1024
#define CDV  192
#define QQ   2
#define KC   2048
#define NVV  4096
#define NVP  4097   // + pad vertex slot

typedef __attribute__((ext_vector_type(16))) __bf16 v16bf;
typedef __attribute__((ext_vector_type(8)))  float  v8f;

// f32 -> bf16 A fragment (contiguous K layout k = 16*half + s; Ap already includes half*16)
__device__ __forceinline__ v16bf load_a_frag(const float* __restrict__ Ap) {
    float4 x0 = *(const float4*)(Ap);
    float4 x1 = *(const float4*)(Ap + 4);
    float4 x2 = *(const float4*)(Ap + 8);
    float4 x3 = *(const float4*)(Ap + 12);
    v16bf a;
    a[0]  = (__bf16)x0.x; a[1]  = (__bf16)x0.y; a[2]  = (__bf16)x0.z; a[3]  = (__bf16)x0.w;
    a[4]  = (__bf16)x1.x; a[5]  = (__bf16)x1.y; a[6]  = (__bf16)x1.z; a[7]  = (__bf16)x1.w;
    a[8]  = (__bf16)x2.x; a[9]  = (__bf16)x2.y; a[10] = (__bf16)x2.z; a[11] = (__bf16)x2.w;
    a[12] = (__bf16)x3.x; a[13] = (__bf16)x3.y; a[14] = (__bf16)x3.z; a[15] = (__bf16)x3.w;
    return a;
}

// ---------------------------------------------------------------------------
// Small utility kernels
// ---------------------------------------------------------------------------
__global__ void k_zero_f32(float* __restrict__ p, size_t n) {
    for (size_t i = blockIdx.x * (size_t)blockDim.x + threadIdx.x; i < n;
         i += (size_t)gridDim.x * blockDim.x)
        p[i] = 0.0f;
}

__global__ void k_set_zero1(float* __restrict__ p) {
    if (threadIdx.x == 0 && blockIdx.x == 0) *p = 0.0f;
}

// f32 [R][C] -> bf16 [C][R]  (transpose so GEMM B-fragments are contiguous in K)
__global__ void k_transpose_cvt(const float* __restrict__ s, __bf16* __restrict__ d,
                                int R, int C) {
    size_t n = (size_t)R * C;
    for (size_t i = blockIdx.x * (size_t)blockDim.x + threadIdx.x; i < n;
         i += (size_t)gridDim.x * blockDim.x) {
        int r = (int)(i / C);
        int c = (int)(i - (size_t)r * C);
        d[(size_t)c * R + r] = (__bf16)s[i];
    }
}

// straight f32 -> bf16 copy (codebooks stay [K][CD]: rows contiguous in CD = K-dim)
__global__ void k_cvt_bf16(const float* __restrict__ s, __bf16* __restrict__ d, size_t n) {
    for (size_t i = blockIdx.x * (size_t)blockDim.x + threadIdx.x; i < n;
         i += (size_t)gridDim.x * blockDim.x)
        d[i] = (__bf16)s[i];
}

// 0.5*||c||^2 per codebook row
__global__ void k_cbnorm(const float* __restrict__ cb, float* __restrict__ cn, int rows) {
    int i = blockIdx.x * blockDim.x + threadIdx.x;
    if (i >= rows) return;
    const float* r = cb + (size_t)i * CDV;
    float s = 0.0f;
    for (int d = 0; d < CDV; ++d) s += r[d] * r[d];
    cn[i] = 0.5f * s;
}

// ---------------------------------------------------------------------------
// bf16 WMMA GEMM:  C[M,N] = A_f32[M,K] * BT_bf16[N,K]^T + bias[N]
// One wave per 16(M) x 64(N) macro-tile: 4 accumulators share one A fragment
// per K-step (4x A reuse).  Software-pipelined: iteration k+32's fragments are
// loaded into fresh registers before the WMMAs consuming iteration k issue,
// so VMEM latency overlaps the matrix pipe and WAR-hazard NOPs are avoided.
// M mult of 16, N mult of 64, K mult of 32 (>= 64).
// Epilogue: wave-private LDS transpose -> coalesced float2 row stores.
// ---------------------------------------------------------------------------
#define LDS_ROW 68   // 64 cols + 4 pad: halves hit disjoint bank groups
__global__ __launch_bounds__(256)
void k_gemm_bf16(const float* __restrict__ A, int lda,
                 const __bf16* __restrict__ BT, int ldb,
                 const float* __restrict__ bias,
                 float* __restrict__ C, int ldc,
                 int Mtiles, int Ntiles4, int K) {
    __shared__ float lds[8][16 * LDS_ROW];
    const int lane = threadIdx.x & 31;
    const int widx = threadIdx.x >> 5;
    const int wave = blockIdx.x * (blockDim.x >> 5) + widx;
    if (wave >= Mtiles * Ntiles4) return;
    const int mt   = wave / Ntiles4;
    const int nt4  = wave - mt * Ntiles4;
    const int half = (lane >> 4) & 1;
    const int l15  = lane & 15;
    const int m  = mt * 16 + l15;
    const int n0 = nt4 * 64;

    const float*  Ap  = A + (size_t)m * lda + (half ? 16 : 0);
    const __bf16* Bp0 = BT + (size_t)(n0 + l15) * ldb + (half ? 16 : 0);
    const __bf16* Bp1 = Bp0 + (size_t)16 * ldb;
    const __bf16* Bp2 = Bp0 + (size_t)32 * ldb;
    const __bf16* Bp3 = Bp0 + (size_t)48 * ldb;

    v8f acc0 = {}, acc1 = {}, acc2 = {}, acc3 = {};

    // prologue: fragments for k0 = 0
    v16bf a  = load_a_frag(Ap);
    v16bf b0 = *(const v16bf*)(Bp0);
    v16bf b1 = *(const v16bf*)(Bp1);
    v16bf b2 = *(const v16bf*)(Bp2);
    v16bf b3 = *(const v16bf*)(Bp3);

    for (int k0 = 32; k0 < K; k0 += 32) {
        __builtin_prefetch(Ap + k0 + 96, 0, 0);   // global_prefetch_b8
        // issue next-iteration loads first (fresh registers)
        v16bf an  = load_a_frag(Ap + k0);
        v16bf bn0 = *(const v16bf*)(Bp0 + k0);
        v16bf bn1 = *(const v16bf*)(Bp1 + k0);
        v16bf bn2 = *(const v16bf*)(Bp2 + k0);
        v16bf bn3 = *(const v16bf*)(Bp3 + k0);
        // consume current fragments
        acc0 = __builtin_amdgcn_wmma_f32_16x16x32_bf16(false, a, false, b0, (short)0, acc0, false, false);
        acc1 = __builtin_amdgcn_wmma_f32_16x16x32_bf16(false, a, false, b1, (short)0, acc1, false, false);
        acc2 = __builtin_amdgcn_wmma_f32_16x16x32_bf16(false, a, false, b2, (short)0, acc2, false, false);
        acc3 = __builtin_amdgcn_wmma_f32_16x16x32_bf16(false, a, false, b3, (short)0, acc3, false, false);
        a = an; b0 = bn0; b1 = bn1; b2 = bn2; b3 = bn3;
    }
    // epilogue K-step
    acc0 = __builtin_amdgcn_wmma_f32_16x16x32_bf16(false, a, false, b0, (short)0, acc0, false, false);
    acc1 = __builtin_amdgcn_wmma_f32_16x16x32_bf16(false, a, false, b1, (short)0, acc1, false, false);
    acc2 = __builtin_amdgcn_wmma_f32_16x16x32_bf16(false, a, false, b2, (short)0, acc2, false, false);
    acc3 = __builtin_amdgcn_wmma_f32_16x16x32_bf16(false, a, false, b3, (short)0, acc3, false, false);

    // wave-private LDS transpose: acc[t] vgpr r holds row (half*8+r), col t*16+l15
    float* sl = lds[widx];
#pragma unroll
    for (int r = 0; r < 8; ++r) {
        const int row = half * 8 + r;
        sl[row * LDS_ROW + 0  + l15] = acc0[r];
        sl[row * LDS_ROW + 16 + l15] = acc1[r];
        sl[row * LDS_ROW + 32 + l15] = acc2[r];
        sl[row * LDS_ROW + 48 + l15] = acc3[r];
    }
    // coalesced float2 stores: lane covers cols n0 + 2*lane, 2*lane+1
    const float bn0 = bias[n0 + lane * 2];
    const float bn1 = bias[n0 + lane * 2 + 1];
    float* Crow = C + (size_t)(mt * 16) * ldc + n0 + lane * 2;
#pragma unroll
    for (int rr = 0; rr < 16; ++rr) {
        float2 v;
        v.x = sl[rr * LDS_ROW + lane * 2]     + bn0;
        v.y = sl[rr * LDS_ROW + lane * 2 + 1] + bn1;
        *(float2*)(Crow + (size_t)rr * ldc) = v;
    }
}

// ---------------------------------------------------------------------------
// Scatter-mean: one 192-thread block per (b, face-corner) row
// ---------------------------------------------------------------------------
__global__ void k_scatter(const float* __restrict__ fe, const int* __restrict__ faces,
                          const unsigned char* __restrict__ mask,
                          float* __restrict__ sum, float* __restrict__ cnt) {
    const int row = blockIdx.x;                // [0, B*NF*3)
    const int b = row / (NF * 3);
    const int t = row - b * (NF * 3);
    const int f = t / 3;
    const int j = t - f * 3;
    const int fm = mask[(size_t)b * NF + f];
    const int v  = fm ? faces[((size_t)b * NF + f) * 3 + j] : NVV;
    const size_t sb = ((size_t)b * NVP + v) * CDV;
    const float* fr = fe + (size_t)row * CDV;
    atomicAdd(&sum[sb + threadIdx.x], fr[threadIdx.x]);
    if (threadIdx.x == 0) atomicAdd(&cnt[(size_t)b * NVP + v], 1.0f);
}

__global__ void k_divide(float* __restrict__ sum, const float* __restrict__ cnt, size_t n) {
    for (size_t i = blockIdx.x * (size_t)blockDim.x + threadIdx.x; i < n;
         i += (size_t)gridDim.x * blockDim.x) {
        size_t row = i / CDV;
        sum[i] /= fmaxf(cnt[row], 1.0f);
    }
}

// ---------------------------------------------------------------------------
// Residual-VQ score GEMM + fused row argmax.
// One wave per 16-row M-tile of the flattened [B*NVP, 192] residual matrix.
// A fragments (6 K-chunks) preloaded once and reused for all 128 N-tiles.
// N-loop software-pipelined: next codebook tile's 6 B fragments are loaded
// while the current tile's 6 WMMAs execute (codebook is L2-resident).
// ---------------------------------------------------------------------------
__global__ __launch_bounds__(256)
void k_vq_argmax(const float* __restrict__ R,      // [Mtot, 192]
                 const __bf16* __restrict__ CB,    // [2048, 192] bf16
                 const float* __restrict__ CN,     // [2048]  0.5*||c||^2
                 int* __restrict__ codes,          // stride QQ, pre-offset by q
                 int Mtot) {
    const int lane = threadIdx.x & 31;
    const int wave = blockIdx.x * (blockDim.x >> 5) + (threadIdx.x >> 5);
    const int m0   = wave * 16;
    if (m0 >= Mtot) return;
    const int half = (lane >> 4) & 1;
    const int l15  = lane & 15;
    const int m    = m0 + l15;

    v16bf a[6];
    if (m < Mtot) {
        const float* Ap = R + (size_t)m * CDV + (half ? 16 : 0);
#pragma unroll
        for (int kk = 0; kk < 6; ++kk)
            a[kk] = load_a_frag(Ap + kk * 32);
    } else {
#pragma unroll
        for (int kk = 0; kk < 6; ++kk)
#pragma unroll
            for (int e = 0; e < 16; ++e) a[kk][e] = (__bf16)0.0f;
    }

    float bestv[8];
    int   besti[8];
#pragma unroll
    for (int r = 0; r < 8; ++r) { bestv[r] = -3.4e38f; besti[r] = 0; }

    const __bf16* Bbase = CB + (half ? 16 : 0);

    // prologue: tile nt = 0
    v16bf b[6];
    {
        const __bf16* Bp = Bbase + (size_t)l15 * CDV;
#pragma unroll
        for (int kk = 0; kk < 6; ++kk) b[kk] = *(const v16bf*)(Bp + kk * 32);
    }

#pragma unroll 1
    for (int nt = 0; nt < KC / 16; ++nt) {
        // issue next tile's loads first (last iteration harmlessly reloads self)
        const int ntn = (nt + 1 < KC / 16) ? nt + 1 : nt;
        const __bf16* Bpn = Bbase + (size_t)(ntn * 16 + l15) * CDV;
        v16bf bn[6];
#pragma unroll
        for (int kk = 0; kk < 6; ++kk) bn[kk] = *(const v16bf*)(Bpn + kk * 32);

        v8f acc = {};
#pragma unroll
        for (int kk = 0; kk < 6; ++kk)
            acc = __builtin_amdgcn_wmma_f32_16x16x32_bf16(false, a[kk], false, b[kk],
                                                          (short)0, acc, false, false);
        const int n = nt * 16 + l15;
        const float cn = CN[n];
#pragma unroll
        for (int r = 0; r < 8; ++r) {
            float v = acc[r] - cn;
            if (v > bestv[r]) { bestv[r] = v; besti[r] = n; }  // strict > keeps first index
        }
#pragma unroll
        for (int kk = 0; kk < 6; ++kk) b[kk] = bn[kk];
    }

    // reduce across the 16 lanes of each half (wave32 shfl, xor masks stay in-half)
#pragma unroll
    for (int s = 1; s < 16; s <<= 1) {
#pragma unroll
        for (int r = 0; r < 8; ++r) {
            float ov = __shfl_xor(bestv[r], s, 32);
            int   oi = __shfl_xor(besti[r], s, 32);
            if (ov > bestv[r] || (ov == bestv[r] && oi < besti[r])) {
                bestv[r] = ov; besti[r] = oi;
            }
        }
    }
    if (l15 == 0) {
#pragma unroll
        for (int r = 0; r < 8; ++r) {
            int mm = m0 + half * 8 + r;
            if (mm < Mtot) codes[(size_t)mm * QQ] = besti[r];
        }
    }
}

// residual -= codebook_f32[q][cid]   (f32 codebook for accuracy)
__global__ void k_vq_update(float* __restrict__ R, const int* __restrict__ codes,
                            const float* __restrict__ cbf32, int q) {
    const int row = blockIdx.x;
    const int cid = codes[(size_t)row * QQ + q];
    const float* c = cbf32 + ((size_t)q * KC + cid) * CDV;
    float* r = R + (size_t)row * CDV;
    for (int d = threadIdx.x; d < CDV; d += blockDim.x) r[d] -= c[d];
}

// commit loss: quantized = avg - residual  =>  (avg - quantized)^2 = residual^2
__global__ void k_commit(const float* __restrict__ R, float* __restrict__ out) {
    const int row = blockIdx.x;
    if (row % NVP == NVP - 1) return;            // pad vertex excluded by vmask
    const float* r = R + (size_t)row * CDV;
    float s = 0.0f;
    for (int d = threadIdx.x; d < CDV; d += 32) { float x = r[d]; s += x * x; }
    for (int sh = 16; sh; sh >>= 1) s += __shfl_xor(s, sh, 32);
    if (threadIdx.x == 0)
        atomicAdd(out, s * (1.0f / (192.0f * (float)(BB * NVV))));
}

// gather quantized (= avg - residual) back onto face corners -> feg [B*NF, 576]
__global__ void k_gather_q(const float* __restrict__ avg, const float* __restrict__ R,
                           const int* __restrict__ faces, const unsigned char* __restrict__ mask,
                           float* __restrict__ feg, size_t total) {
    for (size_t i = blockIdx.x * (size_t)blockDim.x + threadIdx.x; i < total;
         i += (size_t)gridDim.x * blockDim.x) {
        size_t row = i / CDV;
        int d = (int)(i - row * CDV);
        int b = (int)(row / (NF * 3));
        int t = (int)(row - (size_t)b * NF * 3);
        int f = t / 3;
        int j = t - f * 3;
        int v = mask[(size_t)b * NF + f] ? faces[((size_t)b * NF + f) * 3 + j] : NVV;
        size_t s = ((size_t)b * NVP + v) * CDV + d;
        feg[i] = avg[s] - R[s];
    }
}

// codes gathered per face corner, masked with PAD_ID = -1, emitted as float
__global__ void k_codes_out(const int* __restrict__ codes, const int* __restrict__ faces,
                            const unsigned char* __restrict__ mask, float* __restrict__ out,
                            int total) {
    for (int i = blockIdx.x * blockDim.x + threadIdx.x; i < total;
         i += gridDim.x * blockDim.x) {
        int row = i >> 1;               // [B*NF*3]
        int q = i & 1;
        int b = row / (NF * 3);
        int t = row - b * (NF * 3);
        int f = t / 3;
        int j = t - f * 3;
        int fm = mask[(size_t)b * NF + f];
        int c;
        if (fm) {
            int v = faces[((size_t)b * NF + f) * 3 + j];
            c = codes[((size_t)b * NVP + v) * QQ + q];
        } else {
            c = -1;
        }
        out[i] = (float)c;
    }
}

// ---------------------------------------------------------------------------
// Host orchestration
// ---------------------------------------------------------------------------
extern "C" void kernel_launch(void* const* d_in, const int* in_sizes, int n_in,
                              void* d_out, int out_size, void* d_ws, size_t ws_size,
                              hipStream_t stream) {
    const int*           faces      = (const int*)d_in[0];
    const unsigned char* mask       = (const unsigned char*)d_in[1];
    const float*         face_embed = (const float*)d_in[2];
    const float*         W_dc       = (const float*)d_in[3];
    const float*         b_dc       = (const float*)d_in[4];
    const float*         cbf32      = (const float*)d_in[5];
    const float*         W_co       = (const float*)d_in[6];
    const float*         b_co       = (const float*)d_in[7];

    // workspace carve-up (256B aligned)
    char* ws = (char*)d_ws;
    size_t off = 0;
    auto take = [&](size_t bytes) -> char* {
        char* p = ws + off;
        off = (off + bytes + 255) & ~(size_t)255;
        return p;
    };
    __bf16* WdcT   = (__bf16*)take((size_t)DD * 3 * CDV * 2);        // [576][1024]
    __bf16* WcoT   = (__bf16*)take((size_t)3 * CDV * DD * 2);        // [1024][576]
    __bf16* cbbf   = (__bf16*)take((size_t)QQ * KC * CDV * 2);       // [2][2048][192]
    float*  cbnorm = (float*) take((size_t)QQ * KC * 4);
    float*  fe     = (float*) take((size_t)BB * NF * 3 * CDV * 4);   // reused as feg
    float*  avg    = (float*) take((size_t)BB * NVP * CDV * 4);      // sum -> avg (in place)
    float*  cnt    = (float*) take((size_t)BB * NVP * 4);
    float*  resid  = (float*) take((size_t)BB * NVP * CDV * 4);
    int*    codes  = (int*)   take((size_t)BB * NVP * QQ * 4);

    float* out_feo    = (float*)d_out;
    float* out_codes  = out_feo + (size_t)BB * NF * DD;
    float* out_commit = out_codes + (size_t)BB * NF * 3 * QQ;

    const int Mfaces = BB * NF;          // 32768
    const int Mvq    = BB * NVP;         // 32776

    // 1) weight prep: transposed bf16 weights, bf16 codebooks, 0.5*||c||^2
    k_transpose_cvt<<<2304, 256, 0, stream>>>(W_dc, WdcT, DD, 3 * CDV);
    k_transpose_cvt<<<2304, 256, 0, stream>>>(W_co, WcoT, 3 * CDV, DD);
    k_cvt_bf16<<<3072, 256, 0, stream>>>(cbf32, cbbf, (size_t)QQ * KC * CDV);
    k_cbnorm<<<(QQ * KC + 255) / 256, 256, 0, stream>>>(cbf32, cbnorm, QQ * KC);

    // 2) zero scatter accumulators
    k_zero_f32<<<1024, 256, 0, stream>>>(avg, (size_t)BB * NVP * CDV);
    k_zero_f32<<<64, 256, 0, stream>>>(cnt, (size_t)BB * NVP);

    // 3) GEMM1: fe = face_embed @ W_dc + b_dc   [32768x1024]x[1024x576]
    {
        int mt = Mfaces / 16, nt4 = (3 * CDV) / 64;          // 2048 x 9
        k_gemm_bf16<<<(mt * nt4 + 7) / 8, 256, 0, stream>>>(
            face_embed, DD, WdcT, DD, b_dc, fe, 3 * CDV, mt, nt4, DD);
    }

    // 4) scatter-mean into vertex slots
    k_scatter<<<BB * NF * 3, CDV, 0, stream>>>(fe, faces, mask, avg, cnt);
    k_divide<<<2048, 256, 0, stream>>>(avg, cnt, (size_t)BB * NVP * CDV);

    // 5) residual VQ (2 levels); residual starts as avg
    hipMemcpyAsync(resid, avg, (size_t)BB * NVP * CDV * 4,
                   hipMemcpyDeviceToDevice, stream);
    {
        int vqBlocks = ((Mvq + 15) / 16 + 7) / 8;            // waves of 16 rows, 8/block
        k_vq_argmax<<<vqBlocks, 256, 0, stream>>>(resid, cbbf, cbnorm, codes + 0, Mvq);
        k_vq_update<<<Mvq, 64, 0, stream>>>(resid, codes, cbf32, 0);
        k_vq_argmax<<<vqBlocks, 256, 0, stream>>>(resid, cbbf + (size_t)KC * CDV,
                                                  cbnorm + KC, codes + 1, Mvq);
        k_vq_update<<<Mvq, 64, 0, stream>>>(resid, codes, cbf32, 1);
    }

    // 6) commit loss = masked mean of mean_d(residual^2)
    k_set_zero1<<<1, 32, 0, stream>>>(out_commit);
    k_commit<<<Mvq, 32, 0, stream>>>(resid, out_commit);

    // 7) gather quantized (= avg - resid) onto face corners, then GEMM2
    k_gather_q<<<4096, 256, 0, stream>>>(avg, resid, faces, mask, fe,
                                         (size_t)BB * NF * 3 * CDV);
    {
        int mt = Mfaces / 16, nt4 = DD / 64;                 // 2048 x 16
        k_gemm_bf16<<<(mt * nt4 + 7) / 8, 256, 0, stream>>>(
            fe, 3 * CDV, WcoT, 3 * CDV, b_co, out_feo, DD, mt, nt4, 3 * CDV);
    }

    // 8) gathered codes with PAD_ID masking
    k_codes_out<<<768, 256, 0, stream>>>(codes, faces, mask, out_codes,
                                         BB * NF * 3 * QQ);
}
